// Linear2Bit_65446711656920
// MI455X (gfx1250) — matrix operand order repaired and checked by reading, not compiled, not run on hardware
//
#include <hip/hip_runtime.h>
#include <hip/hip_fp16.h>

// ---------------------------------------------------------------------------
// 2-bit quantized linear layer for MI455X (gfx1250, wave32, WMMA).
//   out[m, o] = sum_k x[m,k] * w[o,k] + bias[o],  w = (norm/3)*(2q-3)
// GEMM: M=8192, N=4096, K=4096, bf16 WMMA, f32 accumulate.
//
// Preferred path (needs 96 MB workspace):
//   K1 dequant W -> bf16 (once), K2 convert x -> bf16 (once),
//   K3 GEMM: 3-stage pipeline of global_load_async_to_lds_b128 copies
//            (ASYNCcnt, wait<=4 keeps newest stage in flight across the
//            barrier), 8 v_wmma_f32_16x16x32_bf16 per wave per K-step.
// Fallback (small ws): fused kernel that dequantizes/converts in-loop.
// ---------------------------------------------------------------------------

#define IN_F   4096
#define OUT_F  4096
#define M_TOT  8192          // 4 * 2048 tokens

typedef __attribute__((ext_vector_type(16))) __bf16    v16bf;
typedef __attribute__((ext_vector_type(2)))  __bf16    v2bf;
typedef __attribute__((ext_vector_type(8)))  float     v8f;
typedef __attribute__((ext_vector_type(4)))  unsigned  u32x4;
typedef __attribute__((ext_vector_type(2)))  unsigned  u32x2;
typedef __attribute__((ext_vector_type(4)))  float     f32x4;
typedef __attribute__((ext_vector_type(4)))  int       i32x4;

union FragAB { u32x4 u[2]; v16bf v; };   // 32 bytes: one 16-bf16 WMMA operand

__device__ __forceinline__ unsigned pack_bf16(float lo, float hi) {
  v2bf v = { (__bf16)lo, (__bf16)hi };   // v_cvt_pk_bf16_f32
  return __builtin_bit_cast(unsigned, v);
}

// CDNA5 async global->LDS copy (16B per lane), tracked by ASYNCcnt.
__device__ __forceinline__ void async_copy_b128(unsigned ldsOff, const void* g) {
  asm volatile("global_load_async_to_lds_b128 %0, %1, off"
               :: "v"(ldsOff), "v"(g) : "memory");
}
__device__ __forceinline__ void wait_asynccnt(void) {        // all copies done
  asm volatile("s_wait_asynccnt 0" ::: "memory");
}
__device__ __forceinline__ void wait_asynccnt_keep4(void) {  // newest stage may fly
  asm volatile("s_wait_asynccnt 4" ::: "memory");
}
__device__ __forceinline__ unsigned lds_off(const void* p) {
  return (unsigned)(unsigned long long)p;  // low 32 bits of flat LDS addr = LDS offset
}

// ========================= K1: dequant W -> bf16 ===========================
// One thread per 16-weight group; output is k-contiguous, so group g writes
// 32 contiguous bytes at Wb + g*16 halves.
__global__ __launch_bounds__(256)
void dequant_w_bf16(const int* __restrict__ wq, const __half* __restrict__ wn,
                    unsigned short* __restrict__ Wb) {
  const int g = blockIdx.x * 256 + threadIdx.x;        // 0 .. 1M-1
  const i32x4 q = *(const i32x4*)(wq + (size_t)g * 4);
  const float s    = __half2float(wn[g]) * (1.0f / 3.0f);
  const float twoS = s + s;
  const float m3S  = -3.0f * s;
  unsigned pk[8];
  #pragma unroll
  for (int j = 0; j < 4; ++j) {
    const unsigned by = (unsigned)q[j];
    float w0 = __builtin_fmaf((float)((by      ) & 3u), twoS, m3S);
    float w1 = __builtin_fmaf((float)((by >> 2u) & 3u), twoS, m3S);
    float w2 = __builtin_fmaf((float)((by >> 4u) & 3u), twoS, m3S);
    float w3 = __builtin_fmaf((float)((by >> 6u) & 3u), twoS, m3S);
    pk[2 * j]     = pack_bf16(w0, w1);
    pk[2 * j + 1] = pack_bf16(w2, w3);
  }
  u32x4 lo; lo.x = pk[0]; lo.y = pk[1]; lo.z = pk[2]; lo.w = pk[3];
  u32x4 hi; hi.x = pk[4]; hi.y = pk[5]; hi.z = pk[6]; hi.w = pk[7];
  unsigned short* dst = Wb + (size_t)g * 16;
  *(u32x4*)(dst)     = lo;                              // global_store_b128
  *(u32x4*)(dst + 8) = hi;
}

// ========================= K2: convert x -> bf16 ===========================
__global__ __launch_bounds__(256)
void convert_x_bf16(const float* __restrict__ x, unsigned short* __restrict__ Xb) {
  const size_t gid  = (size_t)blockIdx.x * 256 + threadIdx.x;   // 8 floats each
  const float* src  = x + gid * 8;
  const f32x4 a = *(const f32x4*)(src);
  const f32x4 b = *(const f32x4*)(src + 4);
  u32x4 o;
  o.x = pack_bf16(a.x, a.y); o.y = pack_bf16(a.z, a.w);
  o.z = pack_bf16(b.x, b.y); o.w = pack_bf16(b.z, b.w);
  *(u32x4*)(Xb + gid * 8) = o;                          // global_store_b128
}

// ========================= K3: bf16 GEMM (async, 3-stage) ==================
__global__ __launch_bounds__(256)
void gemm_wmma_bf16(const unsigned short* __restrict__ Xb,   // [M_TOT, IN_F] bf16
                    const unsigned short* __restrict__ Wb,   // [OUT_F, IN_F] bf16
                    const float* __restrict__ bias,
                    float* __restrict__ out) {
  constexpr int BM = 128, BN = 128, BK = 32;
  constexpr int LDA = 40, LDB = 40;        // row pitch in halves (16B aligned)
  constexpr int NK  = IN_F / BK;
  constexpr int NBUF = 3;                  // pipeline depth

  __shared__ __align__(16) unsigned short As[NBUF][BM * LDA];  // 3 x 10 KB
  __shared__ __align__(16) unsigned short Bs[NBUF][BN * LDB];  // 3 x 10 KB

  const int tid   = threadIdx.x;
  const int lane  = tid & 31;
  const int wave  = tid >> 5;
  const int waveM = wave & 3;
  const int waveN = wave >> 2;

  const int nBase = blockIdx.x * BN;
  const int mBase = blockIdx.y * BM;

  // Copy map: tile row = tid>>2 (+64), 16B chunk c = tid&3 (4 per 64B row).
  const int cRow = tid >> 2;               // 0..63
  const int cCh  = tid & 3;                // 0..3

  // Fragment geometry (ISA 7.12.2, 16-bit operands, wave32).
  const int aKh   = (lane >> 4) ? 8 : 0;
  const int bKo   = (lane >> 4) ? 16 : 0;
  const int mLane = lane & 15;

  v8f acc[2][4] = {};

  auto issueCopies = [&](int kt, int buf) {  // 4 async b128 per thread
    const int kBase = kt * BK;
    #pragma unroll
    for (int h = 0; h < 2; ++h) {            // rows cRow and cRow+64
      const int r = cRow + 64 * h;
      async_copy_b128(lds_off(&As[buf][r * LDA + cCh * 8]),
                      Xb + (size_t)(mBase + r) * IN_F + kBase + cCh * 8);
      async_copy_b128(lds_off(&Bs[buf][r * LDB + cCh * 8]),
                      Wb + (size_t)(nBase + r) * IN_F + kBase + cCh * 8);
    }
  };

  // ---------------- prologue: fill stages 0 and 1 ----------------
  issueCopies(0, 0);
  issueCopies(1, 1);
  wait_asynccnt_keep4();                   // stage 0 landed, stage 1 in flight
  __syncthreads();

  // ---------------- main loop ----------------
  int bufK = 0;                            // compute buffer (kt % 3)
  int bufI = 2;                            // issue buffer ((kt+2) % 3)
  #pragma unroll 1
  for (int kt = 0; kt < NK; ++kt) {
    if (kt + 2 < NK) issueCopies(kt + 2, bufI);   // 2 iterations of overlap

    FragAB fa[2], fb[4];
    #pragma unroll
    for (int ms = 0; ms < 2; ++ms) {
      const unsigned short* p = &As[bufK][(waveM * 32 + ms * 16 + mLane) * LDA];
      fa[ms].u[0] = *(const u32x4*)(p + 2 * aKh);
      fa[ms].u[1] = *(const u32x4*)(p + 2 * aKh + 8);
    }
    #pragma unroll
    for (int ns = 0; ns < 4; ++ns) {
      const unsigned short* p = &Bs[bufK][(waveN * 64 + ns * 16 + mLane) * LDB];
      fb[ns].u[0] = *(const u32x4*)(p + bKo);
      fb[ns].u[1] = *(const u32x4*)(p + bKo + 8);
    }

    #pragma unroll
    for (int ms = 0; ms < 2; ++ms)
      #pragma unroll
      for (int ns = 0; ns < 4; ++ns)
        acc[ms][ns] = __builtin_amdgcn_wmma_f32_16x16x32_bf16(
            false, fa[ms].v, false, fb[ns].v, (short)0, acc[ms][ns], false, false);

    // Stage kt+1 (issued last iteration) must have landed; stage kt+2
    // (just issued) may remain in flight across the barrier.
    wait_asynccnt_keep4();
    __syncthreads();

    bufK = (bufK == NBUF - 1) ? 0 : bufK + 1;
    bufI = (bufI == NBUF - 1) ? 0 : bufI + 1;
  }
  wait_asynccnt();                         // drain before LDS reuse/exit

  // ---------------- epilogue: bias + f32 store ----------------
  // C/D layout: VGPR r -> M = r + 8*(lane>=16), N = lane&15
  #pragma unroll
  for (int ms = 0; ms < 2; ++ms) {
    const int mRow = mBase + waveM * 32 + ms * 16 + (lane >> 4) * 8;
    #pragma unroll
    for (int ns = 0; ns < 4; ++ns) {
      const int nCol = nBase + waveN * 64 + ns * 16 + (lane & 15);
      const float bv = bias[nCol];
      float* op = out + (size_t)mRow * OUT_F + nCol;
      #pragma unroll
      for (int r = 0; r < 8; ++r)
        op[(size_t)r * OUT_F] = acc[ms][ns][r] + bv;
    }
  }
}

// ========================= Fallback: fused kernel ==========================
__global__ __launch_bounds__(256)
void linear2bit_fused(const float* __restrict__ x, const int* __restrict__ wq,
                      const __half* __restrict__ wn, const float* __restrict__ bias,
                      float* __restrict__ out) {
  constexpr int BM = 128, BN = 128, BK = 32;
  constexpr int LDA = 40, LDB = 40;
  constexpr int NK  = IN_F / BK;

  __shared__ __align__(16) unsigned short As[2][BM * LDA];
  __shared__ __align__(16) unsigned short Bs[2][BN * LDB];

  const int tid = threadIdx.x, lane = tid & 31, wave = tid >> 5;
  const int waveM = wave & 3, waveN = wave >> 2;
  const int nBase = blockIdx.x * BN, mBase = blockIdx.y * BM;
  const int aCol4 = tid & 7, aRow0 = tid >> 3;
  const int bRow = tid >> 1, bKh = tid & 1;

  f32x4 aStage[4]; i32x4 qStage; float sStage = 0.f;
  const int aKh = (lane >> 4) ? 8 : 0, bKo = (lane >> 4) ? 16 : 0, mLane = lane & 15;
  v8f acc[2][4] = {};

  auto loadStage = [&](int kt) {
    const int kBase = kt * BK;
    const float* xp = x + (size_t)(mBase + aRow0) * IN_F + kBase + aCol4 * 4;
    #pragma unroll
    for (int i = 0; i < 4; ++i)
      aStage[i] = *(const f32x4*)(xp + (size_t)(32 * i) * IN_F);
    const int g = (nBase + bRow) * (IN_F / 16) + (kBase >> 4) + bKh;
    qStage = *(const i32x4*)(wq + (size_t)g * 4);
    sStage = __half2float(wn[g]) * (1.0f / 3.0f);
  };
  auto stageToLds = [&](int buf) {
    #pragma unroll
    for (int i = 0; i < 4; ++i) {
      u32x2 p;
      p.x = pack_bf16(aStage[i].x, aStage[i].y);
      p.y = pack_bf16(aStage[i].z, aStage[i].w);
      *(u32x2*)&As[buf][(aRow0 + 32 * i) * LDA + aCol4 * 4] = p;
    }
    const float twoS = sStage + sStage, m3S = -3.0f * sStage;
    unsigned pk[8];
    #pragma unroll
    for (int j = 0; j < 4; ++j) {
      const unsigned by = (unsigned)qStage[j];
      float w0 = __builtin_fmaf((float)((by      ) & 3u), twoS, m3S);
      float w1 = __builtin_fmaf((float)((by >> 2u) & 3u), twoS, m3S);
      float w2 = __builtin_fmaf((float)((by >> 4u) & 3u), twoS, m3S);
      float w3 = __builtin_fmaf((float)((by >> 6u) & 3u), twoS, m3S);
      pk[2 * j] = pack_bf16(w0, w1); pk[2 * j + 1] = pack_bf16(w2, w3);
    }
    unsigned short* bp = &Bs[buf][bRow * LDB + bKh * 16];
    u32x4 lo; lo.x = pk[0]; lo.y = pk[1]; lo.z = pk[2]; lo.w = pk[3];
    u32x4 hi; hi.x = pk[4]; hi.y = pk[5]; hi.z = pk[6]; hi.w = pk[7];
    *(u32x4*)(bp) = lo; *(u32x4*)(bp + 8) = hi;
  };

  loadStage(0); stageToLds(0); loadStage(1);
  __syncthreads();

  #pragma unroll 1
  for (int kt = 0; kt < NK; ++kt) {
    if (kt + 1 < NK) {
      stageToLds((kt + 1) & 1);
      if (kt + 2 < NK) loadStage(kt + 2);
    }
    const int buf = kt & 1;
    FragAB fa[2], fb[4];
    #pragma unroll
    for (int ms = 0; ms < 2; ++ms) {
      const unsigned short* p = &As[buf][(waveM * 32 + ms * 16 + mLane) * LDA];
      fa[ms].u[0] = *(const u32x4*)(p + 2 * aKh);
      fa[ms].u[1] = *(const u32x4*)(p + 2 * aKh + 8);
    }
    #pragma unroll
    for (int ns = 0; ns < 4; ++ns) {
      const unsigned short* p = &Bs[buf][(waveN * 64 + ns * 16 + mLane) * LDB];
      fb[ns].u[0] = *(const u32x4*)(p + bKo);
      fb[ns].u[1] = *(const u32x4*)(p + bKo + 8);
    }
    #pragma unroll
    for (int ms = 0; ms < 2; ++ms)
      #pragma unroll
      for (int ns = 0; ns < 4; ++ns)
        acc[ms][ns] = __builtin_amdgcn_wmma_f32_16x16x32_bf16(
            false, fa[ms].v, false, fb[ns].v, (short)0, acc[ms][ns], false, false);
    __syncthreads();
  }

  #pragma unroll
  for (int ms = 0; ms < 2; ++ms) {
    const int mRow = mBase + waveM * 32 + ms * 16 + (lane >> 4) * 8;
    #pragma unroll
    for (int ns = 0; ns < 4; ++ns) {
      const int nCol = nBase + waveN * 64 + ns * 16 + (lane & 15);
      const float bv = bias[nCol];
      float* op = out + (size_t)mRow * OUT_F + nCol;
      #pragma unroll
      for (int r = 0; r < 8; ++r)
        op[(size_t)r * OUT_F] = acc[ms][ns][r] + bv;
    }
  }
}

// ================================ host =====================================
extern "C" void kernel_launch(void* const* d_in, const int* in_sizes, int n_in,
                              void* d_out, int out_size, void* d_ws, size_t ws_size,
                              hipStream_t stream) {
  (void)in_sizes; (void)n_in; (void)out_size;
  const float*  x    = (const float*)d_in[0];
  const int*    wq   = (const int*)d_in[1];
  const __half* wn   = (const __half*)d_in[2];
  const float*  bias = (const float*)d_in[3];
  float*        out  = (float*)d_out;

  const size_t needW = (size_t)OUT_F * IN_F * sizeof(unsigned short);  // 32 MB
  const size_t needX = (size_t)M_TOT * IN_F * sizeof(unsigned short);  // 64 MB

  if (ws_size >= needW + needX) {
    unsigned short* Wb = (unsigned short*)d_ws;
    unsigned short* Xb = (unsigned short*)((char*)d_ws + needW);
    dequant_w_bf16<<<(OUT_F * IN_F / 16) / 256, 256, 0, stream>>>(wq, wn, Wb);
    convert_x_bf16<<<(size_t)M_TOT * IN_F / 8 / 256, 256, 0, stream>>>(x, Xb);
    dim3 grid(OUT_F / 128, M_TOT / 128);
    gemm_wmma_bf16<<<grid, 256, 0, stream>>>(Xb, Wb, bias, out);
  } else {
    dim3 grid(OUT_F / 128, M_TOT / 128);
    linear2bit_fused<<<grid, 256, 0, stream>>>(x, wq, wn, bias, out);
  }
}
